// GeneralizedAttention_23845658427761
// MI455X (gfx1250) — matrix operand order repaired and checked by reading, not compiled, not run on hardware
//
#include <hip/hip_runtime.h>

// ---------------------------------------------------------------------------
// Generalized local attention, fused flash-style pipeline for gfx1250 (wave32)
//   K1 pos_embed      : sin/cos tables -> bf16  [h][q][d][k]     (VALU/trans)
//   K2 proj_qkv (x3)  : WMMA GEMM W(256x256) @ x -> q_a/q_g, k, v (bf16)
//   K3 exy            : WMMA GEMM q_g @ pos     -> e_x, e_y (f32)
//   K4 attention      : fused energy + mask + online softmax + P@V (WMMA)
//   K5 final_proj     : WMMA GEMM proj_w @ att + bias, *gamma + residual
// All matrix math: v_wmma_f32_16x16x32_bf16, f32 accumulation.
// A-fragments / contiguous outputs use explicit b128 vector accesses;
// streaming B operands are covered by global_prefetch (per-lane row cover).
// ---------------------------------------------------------------------------

#define C_   256
#define NH_  8
#define D_   32
#define N_   2
#define HT_  64
#define WT_  64
#define HW_  (HT_ * WT_)   // 4096
#define HK_  32
#define WK_  32
#define KVP_ (HK_ * WK_)   // 1024

typedef __attribute__((ext_vector_type(16))) __bf16 bf16x16;
typedef __attribute__((ext_vector_type(8)))  __bf16 bf16x8;
typedef __attribute__((ext_vector_type(8)))  float  f32x8;
typedef __attribute__((ext_vector_type(4)))  float  f32x4;

static __device__ inline f32x8 wmma_bf16(bf16x16 a, bf16x16 b, f32x8 c) {
  // 8 args: (neg_a, A, neg_b, B, c_mod, C, reuse_a, reuse_b)
  return __builtin_amdgcn_wmma_f32_16x16x32_bf16(false, a, false, b, (short)0, c,
                                                 false, false);
}

// A fragment (16x32, 16-bit) in this file's contiguous k order (matches B): lane<16: M=lane, K=0..15;
// lane>=16: M=lane-16, K=16..31.  Two aligned b128 loads per lane.
static __device__ inline bf16x16 load_A_bf16(const __bf16* base, int row_stride,
                                             int lane) {
  const int m  = lane & 15;
  const int kb = (lane & 16) ? 16 : 0;
  const __bf16* p = base + (long)m * row_stride + kb;
  const bf16x8 lo = *(const bf16x8*)(p);
  const bf16x8 hi = *(const bf16x8*)(p + 8);
  return __builtin_shufflevector(lo, hi, 0, 1, 2, 3, 4, 5, 6, 7,
                                 8, 9, 10, 11, 12, 13, 14, 15);
}

// Same fragment built from f32 source (weights), converted to bf16.
static __device__ inline bf16x16 load_A_f32cvt(const float* base, int row_stride,
                                               int lane) {
  const int m  = lane & 15;
  const int kb = (lane & 16) ? 16 : 0;
  const float* p = base + (long)m * row_stride + kb;
  const f32x4* q = (const f32x4*)p;        // 16B aligned by construction
  const f32x4 c0 = q[0], c1 = q[1], c2 = q[2], c3 = q[3];
  bf16x16 a;
#pragma unroll
  for (int j = 0; j < 4; ++j) {
    a[j]      = (__bf16)c0[j];
    a[4 + j]  = (__bf16)c1[j];
    a[8 + j]  = (__bf16)c2[j];
    a[12 + j] = (__bf16)c3[j];
  }
  return a;
}

// B fragment (32x16, 16-bit), same contiguous k order as A: lanes 0-15 hold K=0..15 (col = lane),
// lanes 16-31 hold K=16..31 (col = lane-16). base points at [K][N] row-major.
// Per-element, but each (j, lanes 0..15) is a coalesced 32B segment.
static __device__ inline bf16x16 load_B_bf16(const __bf16* base, long k_stride,
                                             int lane) {
  const int ncol = lane & 15;
  const int kb   = (lane & 16) ? 16 : 0;
  bf16x16 b;
#pragma unroll
  for (int j = 0; j < 16; ++j) b[j] = base[(long)(kb + j) * k_stride + ncol];
  return b;
}

// ---------------------------------------------------------------------------
// K1: positional embedding tables.
// pos[h][q][d][k] = (1/sqrt2) * sum_j sin(a_j)*gw[o][j] + cos(a_j)*gw[o][64+j]
// a_j = (q - 2k) * 1000^(-j/64),  o = h*32+d.   Stored B-ready: [d rows][k cols].
// ---------------------------------------------------------------------------
__global__ void pos_embed_kernel(const float* __restrict__ gwx,
                                 const float* __restrict__ gwy,
                                 __bf16* __restrict__ pos_x,
                                 __bf16* __restrict__ pos_y) {
  const int pair = blockIdx.x;           // q*32 + k
  const int axis = blockIdx.y;           // 0 -> x table, 1 -> y table
  const int q = pair >> 5, k = pair & 31;
  const int o = threadIdx.x;             // output channel 0..255
  const float* gw = axis ? gwy : gwx;
  const float diff = (float)q - 2.0f * (float)k;   // QS=1, KV=2, PM=1
  float acc = 0.0f;
#pragma unroll 4
  for (int j = 0; j < 64; ++j) {
    const float a = diff * __expf(-(float)j * 0.10793368f); // 1000^(-j/64)
    acc = fmaf(__sinf(a), gw[o * 128 + j], acc);
    acc = fmaf(__cosf(a), gw[o * 128 + 64 + j], acc);
  }
  acc *= 0.70710678f;
  const int h = o >> 5, d = o & 31;
  __bf16* dst = axis ? pos_y : pos_x;
  dst[((long)(h * 64 + q) * D_ + d) * (long)WK_ + k] = (__bf16)acc;
}

// ---------------------------------------------------------------------------
// K2: projection GEMM.  Y[o][p] = sum_c W[o][c] * X[n][c][col(p)]
// mode 0 (Q): out0=[n][h][p][d] (+appr_bias), out1 (+geom_bias)   P=4096
// mode 1 (K): out0=[n][h][d][p]                                   P=1024 (subsampled)
// mode 2 (V): out0=[n][h][p][d]                                   P=1024 (subsampled)
// One wave per 16x16 output tile; K loop = 8 WMMAs with next-tile prefetch.
// ---------------------------------------------------------------------------
__global__ __launch_bounds__(256) void proj_qkv_kernel(
    const float* __restrict__ X, const float* __restrict__ W,
    const float* __restrict__ bias0, const float* __restrict__ bias1,
    __bf16* __restrict__ out0, __bf16* __restrict__ out1,
    int P, int mode, int subsample) {
  const int lane = threadIdx.x & 31;
  const int wid  = threadIdx.x >> 5;
  const int tiles = (C_ / 16) * (P / 16);
  const int tile  = blockIdx.x * (blockDim.x >> 5) + wid;
  if (tile >= tiles) return;
  const int n  = blockIdx.y;
  const int pt = tile % (P / 16);
  const int ot = tile / (P / 16);
  const int o0 = ot * 16, p0 = pt * 16;
  const int p  = p0 + (lane & 15);
  // kv grid position p -> spatial offset (stride-2 subsample of 64x64)
  const long colofs = subsample ? (long)((p >> 5) * (2 * WT_) + (p & 31) * 2)
                                : (long)p;
  const float* Xn = X + (long)n * C_ * HW_;
  const int kb   = (lane & 16) ? 16 : 0;
  const int prow = (lane & 15) + kb;      // 32 lanes cover 32 B-tile rows
  f32x8 acc = {};
  for (int k0 = 0; k0 < C_; k0 += 32) {
    if (k0 + 32 < C_)   // cover next B tile: one prefetch per row segment
      __builtin_prefetch(&Xn[(long)(k0 + 32 + prow) * HW_ + colofs], 0, 1);
    bf16x16 a = load_A_f32cvt(W + (long)o0 * C_ + k0, C_, lane);
    bf16x16 b;
#pragma unroll
    for (int j = 0; j < 16; ++j)
      b[j] = (__bf16)Xn[(long)(k0 + kb + j) * HW_ + colofs];
    acc = wmma_bf16(a, b, acc);
  }
  const int m0    = (lane & 16) ? 8 : 0;
  const int h     = o0 >> 5;
  const int dbase = o0 & 31;
  const long nh   = (long)n * NH_ + h;
  if (mode == 0) {
    const long base = (nh * (long)P + p) * D_ + dbase + m0;  // contiguous in v
    bf16x8 r0, r1;
#pragma unroll
    for (int v = 0; v < 8; ++v) {
      const int o = o0 + m0 + v;
      r0[v] = (__bf16)(acc[v] + bias0[o]);
      r1[v] = (__bf16)(acc[v] + bias1[o]);
    }
    *(bf16x8*)(out0 + base) = r0;   // single b128 store
    *(bf16x8*)(out1 + base) = r1;
  } else if (mode == 1) {
#pragma unroll
    for (int v = 0; v < 8; ++v)
      out0[(nh * D_ + dbase + m0 + v) * (long)P + p] = (__bf16)acc[v];
  } else {
    const long base = (nh * (long)P + p) * D_ + dbase + m0;
    bf16x8 r0;
#pragma unroll
    for (int v = 0; v < 8; ++v) r0[v] = (__bf16)acc[v];
    *(bf16x8*)(out0 + base) = r0;
  }
}

// ---------------------------------------------------------------------------
// K3: geometric energy terms.
// e_x[n][h][x][y][kx] = sum_d q_g[n,h,(y,x),d] * pos_x[h,x,kx,d]  (axis=0)
// e_y[n][h][y][x][ky] = sum_d q_g[n,h,(y,x),d] * pos_y[h,y,ky,d]  (axis=1)
// One wave per (axis, n, h, q2, 16-row tile): A=q_g rows, B=pos table, K=32.
// ---------------------------------------------------------------------------
__global__ __launch_bounds__(256) void exy_kernel(
    const __bf16* __restrict__ q_g,
    const __bf16* __restrict__ pos_x, const __bf16* __restrict__ pos_y,
    float* __restrict__ e_x, float* __restrict__ e_y) {
  const int lane = threadIdx.x & 31;
  const int wid  = threadIdx.x >> 5;
  int idx = blockIdx.x * (blockDim.x >> 5) + wid;   // 0..8191
  const int rt   = idx & 3;  idx >>= 2;
  const int q2   = idx & 63; idx >>= 6;
  const int h    = idx & 7;  idx >>= 3;
  const int n    = idx & 1;  idx >>= 1;
  const int axis = idx;                              // 0=e_x, 1=e_y
  const long nh  = (long)n * NH_ + h;
  const int r0   = rt * 16;
  const long apos0   = axis ? ((long)q2 * WT_ + r0) : ((long)r0 * WT_ + q2);
  const int  rstride = (axis ? 1 : WT_) * D_;
  const bf16x16 a = load_A_bf16(q_g + (nh * HW_ + apos0) * D_, rstride, lane);
  const __bf16* Bb = (axis ? pos_y : pos_x) + (long)(h * 64 + q2) * D_ * WK_;
  float* E = axis ? e_y : e_x;
  const int m0 = (lane & 16) ? 8 : 0;
  const int nc = lane & 15;
#pragma unroll
  for (int kt = 0; kt < 2; ++kt) {
    f32x8 acc = {};
    bf16x16 b = load_B_bf16(Bb + kt * 16, WK_, lane);
    acc = wmma_bf16(a, b, acc);
    const int kc = kt * 16 + nc;
#pragma unroll
    for (int v = 0; v < 8; ++v) {
      const int r = r0 + m0 + v;
      E[((nh * 64 + q2) * 64 + r) * (long)WK_ + kc] = acc[v];
    }
  }
}

// ---------------------------------------------------------------------------
// K4: fused local attention (flash-style, online softmax).
// One wave handles 16 queries (fixed y, x0..x0+15). Only ky rows inside the
// local constraint window are visited (~11 of 32): mask computed arithmetically.
// Per ky row: 2 energy WMMAs (32 kx) -> +e_x+e_y -> mask -> online softmax ->
// probs via LDS (C-layout -> A-layout transpose) -> 2 P@V WMMAs (D=32).
// Next ky's K/V tiles are prefetched (32 lanes = 32 row segments each).
// ---------------------------------------------------------------------------
__global__ __launch_bounds__(128) void attention_kernel(
    const __bf16* __restrict__ q_a, const __bf16* __restrict__ pk,
    const __bf16* __restrict__ pv,  const float* __restrict__ e_x,
    const float* __restrict__ e_y,  __bf16* __restrict__ att) {
  __shared__ __align__(16) __bf16 pbuf[4][16 * 32]; // 1KB per wave: probs 16x32
  const int lane = threadIdx.x & 31;
  const int wid  = threadIdx.x >> 5;
  int t = blockIdx.x * 4 + wid;              // 4096 tiles total
  const int x0 = (t & 3) * 16; t >>= 2;
  const int y  = t & 63;       t >>= 6;
  const int h  = t & 7;        t >>= 3;
  const int n  = t;
  const long nh = (long)n * NH_ + h;

  const int ky_lo = (y > 10 ? (y - 10) : 0) >> 1;  // floor((y-10)/2), clamped
  int ky_hi = ((y + 11) >> 1) + 1; if (ky_hi > HK_) ky_hi = HK_;

  const bf16x16 aq =
      load_A_bf16(q_a + (nh * HW_ + (long)y * WT_ + x0) * D_, D_, lane);

  const int m0 = (lane & 16) ? 8 : 0;
  const int nc = lane & 15;
  const int kb = (lane & 16) ? 16 : 0;

  float rmax[8], rsum[8];
  int klo[8], khi[8];
#pragma unroll
  for (int v = 0; v < 8; ++v) {
    rmax[v] = -3.0e38f; rsum[v] = 0.0f;
    const int x = x0 + m0 + v;
    klo[v] = (x > 10 ? (x - 10) : 0) >> 1;
    khi[v] = ((x + 11) >> 1) + 1;
  }
  f32x8 oacc0 = {}, oacc1 = {};

  const float* exb = e_x + (nh * WT_) * (long)HT_ * WK_;       // + (x*HT + y)*WK
  const float* eyb = e_y + ((nh * HT_ + y) * (long)WT_) * HK_; // + x*HK + ky
  const __bf16* kall = pk + nh * (long)D_ * KVP_;
  const __bf16* vall = pv + nh * (long)KVP_ * D_;
  __bf16* pb = &pbuf[wid][0];

  for (int ky = ky_lo; ky < ky_hi; ++ky) {
    if (ky + 1 < ky_hi) {  // prefetch next ky's K and V tiles (one row/lane)
      __builtin_prefetch(&kall[(long)lane * KVP_ + (ky + 1) * WK_], 0, 1);
      __builtin_prefetch(&vall[((long)(ky + 1) * WK_ + lane) * D_], 0, 1);
    }
    // ---- energy: q_a (16x32) @ pk (32 x 32kv) -> two 16x16 f32 tiles
    f32x8 e0 = {}, e1 = {};
    {
      const __bf16* kbase = kall + ky * WK_;
      bf16x16 b0, b1;
#pragma unroll
      for (int j = 0; j < 16; ++j) {
        const __bf16* row = kbase + (long)(kb + j) * KVP_;
        b0[j] = row[nc];
        b1[j] = row[16 + nc];
      }
      e0 = wmma_bf16(aq, b0, e0);
      e1 = wmma_bf16(aq, b1, e1);
    }
    // ---- add geometric terms, apply local-constraint mask
    float tmax[8];
#pragma unroll
    for (int v = 0; v < 8; ++v) {
      const int x = x0 + m0 + v;
      const float  ey  = eyb[(long)x * HK_ + ky];
      const float* exr = exb + ((long)x * HT_ + y) * WK_;
      const int kx0 = nc, kx1 = 16 + nc;
      float a0 = e0[v] + exr[kx0] + ey;
      float a1 = e1[v] + exr[kx1] + ey;
      a0 = (kx0 >= klo[v] && kx0 < khi[v]) ? a0 : -3.0e38f;
      a1 = (kx1 >= klo[v] && kx1 < khi[v]) ? a1 : -3.0e38f;
      e0[v] = a0; e1[v] = a1;
      tmax[v] = fmaxf(a0, a1);
    }
    // ---- row max across the 16 lanes holding this row's columns
#pragma unroll
    for (int s = 1; s < 16; s <<= 1)
#pragma unroll
      for (int v = 0; v < 8; ++v)
        tmax[v] = fmaxf(tmax[v], __shfl_xor(tmax[v], s, 32));
    float scale[8], psum[8];
#pragma unroll
    for (int v = 0; v < 8; ++v) {
      const float nm = fmaxf(rmax[v], tmax[v]);
      scale[v] = __expf(rmax[v] - nm);
      rmax[v]  = nm;
      const float p0 = __expf(e0[v] - nm);
      const float p1 = __expf(e1[v] - nm);
      e0[v] = p0; e1[v] = p1;
      psum[v] = p0 + p1;
    }
#pragma unroll
    for (int s = 1; s < 16; s <<= 1)
#pragma unroll
      for (int v = 0; v < 8; ++v)
        psum[v] += __shfl_xor(psum[v], s, 32);
#pragma unroll
    for (int v = 0; v < 8; ++v) {
      rsum[v]  = rsum[v] * scale[v] + psum[v];
      oacc0[v] *= scale[v];
      oacc1[v] *= scale[v];
    }
    // ---- probs: C-layout -> row-major 16x32 bf16 in LDS (A-layout source)
#pragma unroll
    for (int v = 0; v < 8; ++v) {
      pb[(m0 + v) * 32 + nc]      = (__bf16)e0[v];
      pb[(m0 + v) * 32 + 16 + nc] = (__bf16)e1[v];
    }
    asm volatile("s_wait_dscnt 0x0" ::: "memory");  // per-wave LDS ordering
    const bf16x16 ap = load_A_bf16(pb, 32, lane);   // ds_load_b128 x2
    // ---- P @ V : K = 32 kv rows of this ky, N = D split into two 16-chunks
    const __bf16* vbase = vall + (long)ky * WK_ * D_;
    bf16x16 bv0, bv1;
#pragma unroll
    for (int j = 0; j < 16; ++j) {
      const __bf16* row = vbase + (long)(kb + j) * D_;
      bv0[j] = row[nc];
      bv1[j] = row[16 + nc];
    }
    oacc0 = wmma_bf16(ap, bv0, oacc0);
    oacc1 = wmma_bf16(ap, bv1, oacc1);
    asm volatile("s_wait_dscnt 0x0" ::: "memory");  // loads done before reuse
  }
  // ---- normalize, store att[n][h*32+d][pos] (B-ready for final projection)
  const long posb = (long)y * WT_ + x0;
  bf16x8 r0, r1;
#pragma unroll
  for (int v = 0; v < 8; ++v) {
    const float inv = 1.0f / rsum[v];
    r0[v] = (__bf16)(oacc0[v] * inv);
    r1[v] = (__bf16)(oacc1[v] * inv);
  }
  *(bf16x8*)(att + ((long)n * C_ + h * D_ + nc)      * HW_ + posb + m0) = r0;
  *(bf16x8*)(att + ((long)n * C_ + h * D_ + 16 + nc) * HW_ + posb + m0) = r1;
}

// ---------------------------------------------------------------------------
// K5: output projection + bias + gamma*out + residual (f32 output).
// ---------------------------------------------------------------------------
__global__ __launch_bounds__(256) void final_proj_kernel(
    const __bf16* __restrict__ att, const float* __restrict__ Wp,
    const float* __restrict__ pbias, const float* __restrict__ gamma,
    const float* __restrict__ x, float* __restrict__ out) {
  const int lane = threadIdx.x & 31;
  const int wid  = threadIdx.x >> 5;
  const int tiles = (C_ / 16) * (HW_ / 16);
  const int tile  = blockIdx.x * (blockDim.x >> 5) + wid;
  if (tile >= tiles) return;
  const int n  = blockIdx.y;
  const int pt = tile % (HW_ / 16);
  const int ot = tile / (HW_ / 16);
  const int o0 = ot * 16, p0 = pt * 16;
  const int p  = p0 + (lane & 15);
  const int kb   = (lane & 16) ? 16 : 0;
  const int prow = (lane & 15) + kb;
  const __bf16* An = att + (long)n * C_ * HW_;
  f32x8 acc = {};
  for (int k0 = 0; k0 < C_; k0 += 32) {
    if (k0 + 32 < C_)
      __builtin_prefetch(&An[(long)(k0 + 32 + prow) * HW_ + p], 0, 1);
    bf16x16 a = load_A_f32cvt(Wp + (long)o0 * C_ + k0, C_, lane);
    bf16x16 b;
#pragma unroll
    for (int j = 0; j < 16; ++j)
      b[j] = An[(long)(k0 + kb + j) * HW_ + p];
    acc = wmma_bf16(a, b, acc);
  }
  const float g  = gamma[0];
  const int  m0  = (lane & 16) ? 8 : 0;
#pragma unroll
  for (int v = 0; v < 8; ++v) {
    const int o = o0 + m0 + v;
    const long idx = ((long)n * C_ + o) * HW_ + p;
    out[idx] = g * (acc[v] + pbias[o]) + x[idx];
  }
}

// ---------------------------------------------------------------------------
extern "C" void kernel_launch(void* const* d_in, const int* in_sizes, int n_in,
                              void* d_out, int out_size, void* d_ws, size_t ws_size,
                              hipStream_t stream) {
  const float* x       = (const float*)d_in[0];
  const float* query_w = (const float*)d_in[1];
  const float* key_w   = (const float*)d_in[2];
  const float* value_w = (const float*)d_in[3];
  const float* geomxw  = (const float*)d_in[4];
  const float* geomyw  = (const float*)d_in[5];
  const float* ab      = (const float*)d_in[6];
  const float* gb      = (const float*)d_in[7];
  const float* proj_w  = (const float*)d_in[8];
  const float* proj_b  = (const float*)d_in[9];
  const float* gamma   = (const float*)d_in[10];
  // d_in[11] (mask) unused: local-constraint mask computed arithmetically.
  float* out = (float*)d_out;

  // workspace carve-out (~34 MB total), 256B aligned slices
  char* w = (char*)d_ws;
  auto alloc = [&](size_t bytes) {
    void* p = (void*)w; w += (bytes + 255) & ~(size_t)255; return p;
  };
  __bf16* q_a  = (__bf16*)alloc((size_t)N_ * NH_ * HW_ * D_ * 2);   // 4 MB
  __bf16* q_g  = (__bf16*)alloc((size_t)N_ * NH_ * HW_ * D_ * 2);   // 4 MB
  __bf16* pk   = (__bf16*)alloc((size_t)N_ * NH_ * D_ * KVP_ * 2);  // 1 MB
  __bf16* pv   = (__bf16*)alloc((size_t)N_ * NH_ * KVP_ * D_ * 2);  // 1 MB
  __bf16* posx = (__bf16*)alloc((size_t)NH_ * WT_ * D_ * WK_ * 2);  // 1 MB
  __bf16* posy = (__bf16*)alloc((size_t)NH_ * HT_ * D_ * HK_ * 2);  // 1 MB
  float*  e_x  = (float*)alloc((size_t)N_ * NH_ * WT_ * HT_ * WK_ * 4); // 8 MB
  float*  e_y  = (float*)alloc((size_t)N_ * NH_ * HT_ * WT_ * HK_ * 4); // 8 MB
  __bf16* att  = (__bf16*)alloc((size_t)N_ * C_ * HW_ * 2);         // 4 MB

  // K1: pos tables (2048 (q,k) pairs x 2 axes, 256 threads over channels)
  pos_embed_kernel<<<dim3(WT_ * WK_, 2), dim3(256), 0, stream>>>(
      geomxw, geomyw, posx, posy);

  // K2: projections (one wave per 16x16 tile, 8 waves/block)
  proj_qkv_kernel<<<dim3((16 * (HW_ / 16)) / 8, N_), dim3(256), 0, stream>>>(
      x, query_w, ab, gb, q_a, q_g, HW_, 0, 0);
  proj_qkv_kernel<<<dim3((16 * (KVP_ / 16)) / 8, N_), dim3(256), 0, stream>>>(
      x, key_w, nullptr, nullptr, pk, nullptr, KVP_, 1, 1);
  proj_qkv_kernel<<<dim3((16 * (KVP_ / 16)) / 8, N_), dim3(256), 0, stream>>>(
      x, value_w, nullptr, nullptr, pv, nullptr, KVP_, 2, 1);

  // K3: e_x / e_y GEMMs (8192 waves)
  exy_kernel<<<dim3(8192 / 8), dim3(256), 0, stream>>>(q_g, posx, posy, e_x, e_y);

  // K4: fused attention (4096 waves, 4 per block, 4 KB LDS)
  attention_kernel<<<dim3(4096 / 4), dim3(128), 0, stream>>>(
      q_a, pk, pv, e_x, e_y, att);

  // K5: output projection + residual
  final_proj_kernel<<<dim3((16 * (HW_ / 16)) / 8, N_), dim3(256), 0, stream>>>(
      att, proj_w, proj_b, gamma, x, out);
}
